// quant_custom_mini_resnet_83099027243745
// MI455X (gfx1250) — 8-bit matrix operand order repaired and checked by reading, compiled, not run on hardware
//
#include <hip/hip_runtime.h>
#include <hip/hip_bf16.h>

typedef signed char i8;
typedef int v8i __attribute__((ext_vector_type(8)));
typedef _Float16 h8 __attribute__((ext_vector_type(8)));
typedef i8 c8 __attribute__((ext_vector_type(8)));

constexpr size_t SC   = 0;
constexpr size_t WQ1  = 256;
constexpr size_t WQ2  = WQ1 + 2048;
constexpr size_t WQ3  = WQ2 + 20480;
constexpr size_t WQ4  = WQ3 + 1024;
constexpr size_t WQ5  = WQ4 + 3072;
constexpr size_t WLQ  = WQ5 + 4096;
constexpr size_t BQ   = WLQ + 768;
constexpr size_t QX   = BQ + 1536;
constexpr size_t SK1  = QX + 4194304;
constexpr size_t SK2  = SK1 + 1048576;
constexpr size_t TBUF = SK2 + 4194304;
constexpr size_t AQA  = TBUF + 67108864;
constexpr size_t AQB  = AQA + 33554432;

__global__ void zero_scalars(unsigned* s) { if (threadIdx.x < 32) s[threadIdx.x] = 0u; }

__global__ __launch_bounds__(256) void absmax_f32(const float* __restrict__ x, size_t n,
                                                  unsigned* __restrict__ slot) {
    __shared__ float red[256];
    float lm = 0.f;
    for (size_t i = (size_t)blockIdx.x * blockDim.x + threadIdx.x; i < n;
         i += (size_t)gridDim.x * blockDim.x)
        lm = fmaxf(lm, fabsf(x[i]));
    red[threadIdx.x] = lm; __syncthreads();
    for (int s = 128; s > 0; s >>= 1) {
        if ((int)threadIdx.x < s) red[threadIdx.x] = fmaxf(red[threadIdx.x], red[threadIdx.x + s]);
        __syncthreads();
    }
    if (threadIdx.x == 0) atomicMax(slot, __float_as_uint(red[0]));
}

__global__ __launch_bounds__(256) void quant_input(const float* __restrict__ x,
                                                   i8* __restrict__ q,
                                                   const float* __restrict__ scal, size_t n4) {
    float s = 127.f / fmaxf(scal[0], 1e-20f);
    for (size_t i = (size_t)blockIdx.x * blockDim.x + threadIdx.x; i < n4;
         i += (size_t)gridDim.x * blockDim.x) {
        float4 xv = *(const float4*)(x + i * 4);
        int out = 0;
        float vv[4] = {xv.x, xv.y, xv.z, xv.w};
#pragma unroll
        for (int j = 0; j < 4; ++j) {
            int v = __float2int_rn(vv[j] * s);
            v = v > 127 ? 127 : (v < -127 ? -127 : v);
            out |= (v & 0xff) << (8 * j);
        }
        *(int*)(q + i * 4) = out;
    }
}

__global__ __launch_bounds__(256) void quant_act(const _Float16* __restrict__ t,
                                                 i8* __restrict__ q,
                                                 const float* __restrict__ scal, int slot, size_t n8) {
    float s = 127.f / fmaxf(scal[slot], 1e-20f);
    for (size_t i = (size_t)blockIdx.x * blockDim.x + threadIdx.x; i < n8;
         i += (size_t)gridDim.x * blockDim.x) {
        h8 hv = *(const h8*)(t + i * 8);
        c8 ov;
#pragma unroll
        for (int j = 0; j < 8; ++j) {
            int v = __float2int_rn((float)hv[j] * s);
            v = v > 127 ? 127 : (v < 0 ? 0 : v);
            ov[j] = (i8)v;
        }
        *(c8*)(q + i * 8) = ov;
    }
}

__global__ __launch_bounds__(256) void maxpool_x(const i8* __restrict__ qx,
                                                 float* __restrict__ sk,
                                                 const float* __restrict__ scal, int Bn) {
    int idx = blockIdx.x * blockDim.x + threadIdx.x;
    if (idx >= Bn * 64) return;
    int b = idx >> 6, rem = idx & 63, oy = rem >> 3, ox = rem & 7;
    int m = -128;
    for (int wy = 0; wy < 5; ++wy) {
        int iy = oy * 4 - 2 + wy; if (iy < 0 || iy >= 32) continue;
        for (int wx = 0; wx < 5; ++wx) {
            int ix = ox * 4 - 2 + wx; if (ix < 0 || ix >= 32) continue;
            int v = qx[(b * 32 + iy) * 32 + ix];
            m = v > m ? v : m;
        }
    }
    sk[idx] = (float)m * (scal[0] * (1.f / 127.f));
}

__global__ __launch_bounds__(256) void maxpool_a2(const i8* __restrict__ a2,
                                                  i8* __restrict__ sk, int Bn) {
    int idx = blockIdx.x * blockDim.x + threadIdx.x;
    if (idx >= Bn * 1024) return;
    int c = idx & 63, r = idx >> 6;
    int ox = r & 3; r >>= 2; int oy = r & 3; int b = r >> 2;
    int m = -128;
    for (int wy = 0; wy < 3; ++wy) {
        int iy = oy * 2 - 1 + wy; if (iy < 0 || iy >= 8) continue;
        for (int wx = 0; wx < 3; ++wx) {
            int ix = ox * 2 - 1 + wx; if (ix < 0 || ix >= 8) continue;
            int v = a2[((size_t)(b * 8 + iy) * 8 + ix) * 64 + c];
            m = v > m ? v : m;
        }
    }
    sk[idx] = (i8)m;
}

__global__ __launch_bounds__(256) void prep_weights(
    const float* w1, const float* b1, const float* w2, const float* b2,
    const float* w3, const float* b3, const float* w4, const float* b4,
    const float* w5, const float* b5, const float* wl, const float* bl,
    float* __restrict__ scalF, char* __restrict__ ws) {
    __shared__ float rw[256], rb[256];
    __shared__ float shW, shB;
    const float *w, *b; int N, IC, KHW, Kpad, slotW; i8* wq; float* bqd;
    switch (blockIdx.x) {
    case 0: w=w1; b=b1; N=32; IC=1;  KHW=9; Kpad=64;  wq=(i8*)(ws+WQ1); bqd=(float*)(ws+BQ+0*256); slotW=1; break;
    case 1: w=w2; b=b2; N=64; IC=32; KHW=9; Kpad=320; wq=(i8*)(ws+WQ2); bqd=(float*)(ws+BQ+1*256); slotW=2; break;
    case 2: w=w3; b=b3; N=16; IC=64; KHW=1; Kpad=64;  wq=(i8*)(ws+WQ3); bqd=(float*)(ws+BQ+2*256); slotW=3; break;
    case 3: w=w4; b=b4; N=16; IC=16; KHW=9; Kpad=192; wq=(i8*)(ws+WQ4); bqd=(float*)(ws+BQ+3*256); slotW=4; break;
    case 4: w=w5; b=b5; N=64; IC=16; KHW=1; Kpad=64;  wq=(i8*)(ws+WQ5); bqd=(float*)(ws+BQ+4*256); slotW=5; break;
    default: w=wl; b=bl; N=10; IC=64; KHW=1; Kpad=64; wq=(i8*)(ws+WLQ); bqd=(float*)(ws+BQ+5*256); slotW=6; break;
    }
    const int K = IC * KHW;
    const int tid = threadIdx.x;
    float lm = 0.f, lb = 0.f;
    for (int i = tid; i < N * K; i += 256) lm = fmaxf(lm, fabsf(w[i]));
    for (int i = tid; i < N; i += 256)     lb = fmaxf(lb, fabsf(b[i]));
    rw[tid] = lm; rb[tid] = lb; __syncthreads();
    for (int s = 128; s > 0; s >>= 1) {
        if (tid < s) { rw[tid] = fmaxf(rw[tid], rw[tid + s]); rb[tid] = fmaxf(rb[tid], rb[tid + s]); }
        __syncthreads();
    }
    if (tid == 0) {
        float mw = fmaxf(rw[0], 1e-20f), mb = fmaxf(rb[0], 1e-20f);
        scalF[slotW] = mw;
        shW = 127.f / mw; shB = 32767.f / mb;
    }
    __syncthreads();
    float sw = shW, sb = shB;
    for (int i = tid; i < N * Kpad; i += 256) {
        int n = i / Kpad, k = i - n * Kpad, v = 0;
        if (k < K) {
            int tap = k / IC, c = k - tap * IC;
            float val = w[(n * IC + c) * KHW + tap];
            v = __float2int_rn(val * sw);
            v = v > 127 ? 127 : (v < -127 ? -127 : v);
        }
        wq[i] = (i8)v;
    }
    for (int i = tid; i < N; i += 256) {
        float q = rintf(b[i] * sb);
        q = fminf(fmaxf(q, -32767.f), 32767.f);
        bqd[i] = q / sb;
    }
}

template <int IH, int IW, int IC, int OH, int OW, int OC,
          int KH, int KW, int STR, int PAD, int K, int KPAD, int SKIPMODE>
__global__ __launch_bounds__(128) void conv_wmma(
    const i8* __restrict__ aq, const i8* __restrict__ wq, const float* __restrict__ bqd,
    const float* __restrict__ scal, int slotA, int slotW,
    _Float16* __restrict__ tout, unsigned* __restrict__ maxOut,
    const float* __restrict__ skipF, const i8* __restrict__ skipQ, int slotSkip) {
    constexpr int HW   = OH * OW;
    constexpr int NSUB = OC / 16;
    constexpr int NK   = KPAD / 64;
    __shared__ __align__(16) i8 ldsA[4][16][KPAD];
    __shared__ float smax[128];
    const int wid  = threadIdx.x >> 5;
    const int lane = threadIdx.x & 31;
    const int h    = lane >> 4;
    const int l16  = lane & 15;
    const int tileBase = (blockIdx.x * 4 + wid) * 16;

    i8 (*At)[KPAD] = ldsA[wid];
    if constexpr ((IC & 3) == 0) {
        for (int idx = lane; idx < 16 * (KPAD / 4); idx += 32) {
            int r = idx / (KPAD / 4);
            int k = (idx - r * (KPAD / 4)) * 4;
            int v = 0;
            if (k < K) {
                int row = tileBase + r;
                int bb = row / HW; int rem = row - bb * HW;
                int oy = rem / OW, ox = rem - oy * OW;
                int tap = k / IC, c = k - tap * IC;
                int ky = tap / KW, kx = tap - ky * KW;
                int iy = oy * STR - PAD + ky, ix = ox * STR - PAD + kx;
                if (iy >= 0 && iy < IH && ix >= 0 && ix < IW)
                    v = *(const int*)(aq + ((size_t)(bb * IH + iy) * IW + ix) * IC + c);
            }
            *(int*)&At[r][k] = v;
        }
    } else {
        for (int idx = lane; idx < 16 * KPAD; idx += 32) {
            int r = idx / KPAD, k = idx - r * KPAD;
            i8 v = 0;
            if (k < K) {
                int row = tileBase + r;
                int bb = row / HW; int rem = row - bb * HW;
                int oy = rem / OW, ox = rem - oy * OW;
                int tap = k / IC, c = k - tap * IC;
                int ky = tap / KW, kx = tap - ky * KW;
                int iy = oy * STR - PAD + ky, ix = ox * STR - PAD + kx;
                if (iy >= 0 && iy < IH && ix >= 0 && ix < IW)
                    v = aq[((size_t)(bb * IH + iy) * IW + ix) * IC + c];
            }
            At[r][k] = v;
        }
    }
    __syncthreads();

    v8i acc[NSUB];
    v8i zero = {0,0,0,0,0,0,0,0};
#pragma unroll
    for (int i = 0; i < NSUB; ++i) acc[i] = zero;

#pragma unroll
    for (int kc = 0; kc < NK; ++kc) {
        v8i va;
        const i8* arow = &At[l16][kc * 64 + h * 16];
#pragma unroll
        for (int g = 0; g < 4; ++g) {
            long long d = *(const long long*)(arow + (g >> 1) * 32 + (g & 1) * 8);
            va[2 * g]     = (int)(unsigned)(d & 0xffffffffLL);
            va[2 * g + 1] = (int)(unsigned)(((unsigned long long)d) >> 32);
        }
#pragma unroll
        for (int ns = 0; ns < NSUB; ++ns) {
            const i8* wrow = wq + (size_t)(ns * 16 + l16) * KPAD + kc * 64 + h * 16;
            int4 p0 = *(const int4*)(wrow);
            int4 p1 = *(const int4*)(wrow + 32);
            v8i vb;
            vb[0] = p0.x; vb[1] = p0.y; vb[2] = p0.z; vb[3] = p0.w;
            vb[4] = p1.x; vb[5] = p1.y; vb[6] = p1.z; vb[7] = p1.w;
            acc[ns] = __builtin_amdgcn_wmma_i32_16x16x64_iu8(
                true, va, true, vb, acc[ns], false, false);
        }
    }

    const float descale = (scal[slotA] * scal[slotW]) * (1.0f / (127.f * 127.f));
    const float skDeq   = (SKIPMODE == 2) ? scal[slotSkip] * (1.0f / 127.f) : 0.f;
    float lmax = 0.f;
#pragma unroll
    for (int r = 0; r < 8; ++r) {
        int row = tileBase + r + h * 8;
        size_t obase = (size_t)row * OC;
        float skipv = 0.f;
        if constexpr (SKIPMODE == 1) skipv = skipF[row];
#pragma unroll
        for (int ns = 0; ns < NSUB; ++ns) {
            int n = ns * 16 + l16;
            float y = (float)acc[ns][r] * descale + bqd[n] + skipv;
            if constexpr (SKIPMODE == 2)
                y += (float)skipQ[obase + n] * skDeq;
            float t = fminf(fmaxf(y, 0.f), 10.f);
            tout[obase + n] = (_Float16)t;
            lmax = fmaxf(lmax, t);
        }
    }
    smax[threadIdx.x] = lmax; __syncthreads();
    for (int s = 64; s > 0; s >>= 1) {
        if ((int)threadIdx.x < s) smax[threadIdx.x] = fmaxf(smax[threadIdx.x], smax[threadIdx.x + s]);
        __syncthreads();
    }
    if (threadIdx.x == 0) atomicMax(maxOut, __float_as_uint(smax[0]));
}

__global__ __launch_bounds__(64) void head(const i8* __restrict__ a5, const i8* __restrict__ wlq,
                                           const float* __restrict__ blq,
                                           const float* __restrict__ scal,
                                           float* __restrict__ out, int Bn) {
    __shared__ float mch[64];
    int b = blockIdx.x, c = threadIdx.x;
    if (b >= Bn) return;
    int s = 0;
    for (int p = 0; p < 16; ++p) s += a5[((size_t)b * 16 + p) * 64 + c];
    mch[c] = (float)s * (scal[12] * (1.f / 127.f)) * (1.f / 16.f);
    __syncthreads();
    if (c < 10) {
        float wlinv = scal[6] * (1.f / 127.f);
        float acc = 0.f;
        for (int k = 0; k < 64; ++k) acc += mch[k] * (float)wlq[c * 64 + k];
        out[b * 10 + c] = acc * wlinv + blq[c];
    }
}

extern "C" void kernel_launch(void* const* d_in, const int* in_sizes, int n_in,
                              void* d_out, int out_size, void* d_ws, size_t ws_size,
                              hipStream_t stream) {
    (void)in_sizes; (void)n_in; (void)out_size; (void)ws_size;
    const float* x  = (const float*)d_in[0];
    const float* w1 = (const float*)d_in[1];  const float* b1 = (const float*)d_in[2];
    const float* w2 = (const float*)d_in[3];  const float* b2 = (const float*)d_in[4];
    const float* w3 = (const float*)d_in[5];  const float* b3 = (const float*)d_in[6];
    const float* w4 = (const float*)d_in[7];  const float* b4 = (const float*)d_in[8];
    const float* w5 = (const float*)d_in[9];  const float* b5 = (const float*)d_in[10];
    const float* wl = (const float*)d_in[11]; const float* bl = (const float*)d_in[12];

    char* ws = (char*)d_ws;
    unsigned* scalU = (unsigned*)(ws + SC);
    float*    scalF = (float*)(ws + SC);
    i8* wq1 = (i8*)(ws + WQ1); i8* wq2 = (i8*)(ws + WQ2); i8* wq3 = (i8*)(ws + WQ3);
    i8* wq4 = (i8*)(ws + WQ4); i8* wq5 = (i8*)(ws + WQ5); i8* wlq = (i8*)(ws + WLQ);
    float* bq1 = (float*)(ws + BQ + 0 * 256); float* bq2 = (float*)(ws + BQ + 1 * 256);
    float* bq3 = (float*)(ws + BQ + 2 * 256); float* bq4 = (float*)(ws + BQ + 3 * 256);
    float* bq5 = (float*)(ws + BQ + 4 * 256); float* blq = (float*)(ws + BQ + 5 * 256);
    i8* qx  = (i8*)(ws + QX);
    float* sk1 = (float*)(ws + SK1);
    i8* sk2 = (i8*)(ws + SK2);
    _Float16* tb = (_Float16*)(ws + TBUF);
    i8* aqA = (i8*)(ws + AQA);
    i8* aqB = (i8*)(ws + AQB);
    float* out = (float*)d_out;
    const int B = 4096;

    zero_scalars<<<1, 32, 0, stream>>>(scalU);
    prep_weights<<<6, 256, 0, stream>>>(w1,b1,w2,b2,w3,b3,w4,b4,w5,b5,wl,bl, scalF, ws);
    absmax_f32<<<1024, 256, 0, stream>>>(x, (size_t)B * 1024, scalU + 0);
    quant_input<<<1024, 256, 0, stream>>>(x, qx, scalF, (size_t)B * 1024 / 4);
    maxpool_x<<<(B * 64 + 255) / 256, 256, 0, stream>>>(qx, sk1, scalF, B);

    conv_wmma<32,32,1, 16,16,32, 3,3,2,1, 9,64, 0><<<(B * 16 * 16) / 64, 128, 0, stream>>>(
        qx, wq1, bq1, scalF, 0, 1, tb, scalU + 8, nullptr, nullptr, 0);
    quant_act<<<2048, 256, 0, stream>>>(tb, aqA, scalF, 8, (size_t)B * 16 * 16 * 32 / 8);

    conv_wmma<16,16,32, 8,8,64, 3,3,2,1, 288,320, 1><<<(B * 8 * 8) / 64, 128, 0, stream>>>(
        aqA, wq2, bq2, scalF, 8, 2, tb, scalU + 9, sk1, nullptr, 0);
    quant_act<<<1024, 256, 0, stream>>>(tb, aqB, scalF, 9, (size_t)B * 8 * 8 * 64 / 8);
    maxpool_a2<<<(B * 1024 + 255) / 256, 256, 0, stream>>>(aqB, sk2, B);

    conv_wmma<8,8,64, 8,8,16, 1,1,1,0, 64,64, 0><<<(B * 8 * 8) / 64, 128, 0, stream>>>(
        aqB, wq3, bq3, scalF, 9, 3, tb, scalU + 10, nullptr, nullptr, 0);
    quant_act<<<256, 256, 0, stream>>>(tb, aqA, scalF, 10, (size_t)B * 8 * 8 * 16 / 8);

    conv_wmma<8,8,16, 4,4,16, 3,3,2,1, 144,192, 0><<<(B * 4 * 4) / 64, 128, 0, stream>>>(
        aqA, wq4, bq4, scalF, 10, 4, tb, scalU + 11, nullptr, nullptr, 0);
    quant_act<<<64, 256, 0, stream>>>(tb, aqB, scalF, 11, (size_t)B * 4 * 4 * 16 / 8);

    conv_wmma<4,4,16, 4,4,64, 1,1,1,0, 16,64, 2><<<(B * 4 * 4) / 64, 128, 0, stream>>>(
        aqB, wq5, bq5, scalF, 11, 5, tb, scalU + 12, nullptr, sk2, 9);
    quant_act<<<256, 256, 0, stream>>>(tb, aqA, scalF, 12, (size_t)B * 4 * 4 * 64 / 8);

    head<<<B, 64, 0, stream>>>(aqA, wlq, blq, scalF, out, B);
}
